// SelfAttentionModel_39848706573196
// MI455X (gfx1250) — matrix operand order repaired and checked by reading, not compiled, not run on hardware
//
#include <hip/hip_runtime.h>
#include <hip/hip_bf16.h>

typedef __attribute__((ext_vector_type(16))) _Float16 v16h;
typedef __attribute__((ext_vector_type(8)))  float    v8f;
typedef int v4i __attribute__((vector_size(16)));

union HFrag {
    v16h     v;
    _Float16 h[16];
    uint4    q[2];
};

union PPack {
    _Float16 h[8];
    unsigned u[4];
    uint4    q;
};

constexpr int Bc = 2, Sc = 2048, Hc = 16, Dh = 64;

// Cross-half (xor-16) exchange in wave32 as a pure VALU op (no LDS traffic).
__device__ __forceinline__ unsigned permx16u(unsigned x) {
    return __builtin_amdgcn_permlanex16(x, x, 0x76543210, 0xFEDCBA98u, false, false);
}
__device__ __forceinline__ float xor16f(float x) {
    return __uint_as_float(permx16u(__float_as_uint(x)));
}

__device__ __forceinline__ v8f wmma_f16(const HFrag& a, const HFrag& b, v8f c) {
    return __builtin_amdgcn_wmma_f32_16x16x32_f16(false, a.v, false, b.v,
                                                  (short)0, c, false, false);
}

// ---- CDNA5 async global->LDS copy (ASYNCcnt-tracked), with sync fallback ----
#if __has_builtin(__builtin_amdgcn_global_load_async_to_lds_b128) && \
    __has_builtin(__builtin_amdgcn_s_wait_asynccnt)
#define USE_ASYNC_LDS 1
#else
#define USE_ASYNC_LDS 0
#endif

typedef __attribute__((address_space(1))) v4i* g4p;   // global int4*
typedef __attribute__((address_space(3))) v4i* l4p;   // LDS int4*

__device__ __forceinline__ void async_cp16(void* lds_dst, const void* gsrc) {
#if USE_ASYNC_LDS
    __builtin_amdgcn_global_load_async_to_lds_b128((g4p)gsrc, (l4p)lds_dst, 0, 0);
#else
    *(uint4*)lds_dst = *(const uint4*)gsrc;
#endif
}
__device__ __forceinline__ void wait_async() {
#if USE_ASYNC_LDS
    __builtin_amdgcn_s_wait_asynccnt(0);
#endif
}

// ---------------------------------------------------------------------------
// Kernel 1: fused Q/K/V projection (fp32 math, f16 output), Q pre-scaled by
// 1/sqrt(D). Q,K stored [B,H,S,D]; V stored transposed [B,H,D,S] so the
// attention kernel can load V^T A-fragments contiguously.
// ---------------------------------------------------------------------------
__global__ __launch_bounds__(256) void proj_kernel(
    const float* __restrict__ qin, const float* __restrict__ kin,
    const float* __restrict__ vin,
    const float* __restrict__ Wq, const float* __restrict__ bq,
    const float* __restrict__ Wk, const float* __restrict__ bk,
    const float* __restrict__ Wv, const float* __restrict__ bv,
    const float* __restrict__ inv_scale,
    _Float16* __restrict__ qf, _Float16* __restrict__ kf,
    _Float16* __restrict__ vt)
{
    __shared__ __align__(16) float xs[3][16][Dh];   // 12 KB input rows
    __shared__ float Ws[3][Dh][65];                 // padded weights (conflict-free)

    const int t  = threadIdx.x;
    const int h  = blockIdx.y;
    const int b  = blockIdx.z;
    const int s0 = blockIdx.x * 16;
    const int bh = b * Hc + h;

    // stage 16 input rows per tensor (coalesced float4)
    {
        const float* srcs[3] = {qin, kin, vin};
        const int row = t >> 4, c4 = (t & 15) * 4;
        for (int j = 0; j < 3; ++j) {
            float4 val = *(const float4*)(srcs[j] +
                (((size_t)(b * Sc + s0 + row)) * Hc + h) * Dh + c4);
            *(float4*)&xs[j][row][c4] = val;
        }
        // stage weights (64x64 each) with padding -> scalar stores
        const float* wsrc[3] = {Wq, Wk, Wv};
        for (int j = 0; j < 3; ++j)
            for (int p = 0; p < 4; ++p) {
                int idx = p * 256 + t;               // 1024 groups of 4 floats
                int wr = idx >> 4, wc = (idx & 15) * 4;
                float4 w4 = *(const float4*)(wsrc[j] + wr * Dh + wc);
                Ws[j][wr][wc + 0] = w4.x;
                Ws[j][wr][wc + 1] = w4.y;
                Ws[j][wr][wc + 2] = w4.z;
                Ws[j][wr][wc + 3] = w4.w;
            }
    }
    __syncthreads();

    const int   e   = t & 63;
    const int   r0  = (t >> 6) * 4;
    const float qsc = 1.0f / inv_scale[0];
    const float bqe = bq[e], bke = bk[e], bve = bv[e];

    for (int rr = 0; rr < 4; ++rr) {
        const int row = r0 + rr;
        float aq = 0.f, ak = 0.f, av = 0.f;
#pragma unroll 8
        for (int d = 0; d < Dh; ++d) {
            aq += Ws[0][e][d] * xs[0][row][d];
            ak += Ws[1][e][d] * xs[1][row][d];
            av += Ws[2][e][d] * xs[2][row][d];
        }
        const int s = s0 + row;
        qf[((size_t)bh * Sc + s) * Dh + e] = (_Float16)((aq + bqe) * qsc);
        kf[((size_t)bh * Sc + s) * Dh + e] = (_Float16)(ak + bke);
        vt[((size_t)bh * Dh + e) * Sc + s] = (_Float16)(av + bve);
    }
}

// ---------------------------------------------------------------------------
// Kernel 2: flash attention with transposed scores, 64 keys per stage,
// double-buffered LDS with async global->LDS pipeline (1 barrier / stage).
//   S^T[key,q] = K_tile(16x64) x Q^T(64x16)   -> 8 wmma per stage (4 tiles)
//   O^T[d,q]  += V^T_tile(16x64) x P^T(64x16) -> 8 wmma per stage (4 d-tiles)
// Block = 256 threads = 8 waves; each wave owns one 16-query tile.
// ---------------------------------------------------------------------------
__global__ __launch_bounds__(256) void attn_kernel(
    const _Float16* __restrict__ qf, const _Float16* __restrict__ kf,
    const _Float16* __restrict__ vt, const float* __restrict__ mask,
    float* __restrict__ out)
{
    __shared__ __align__(16) _Float16 lds_k[2][64][72];  // 64 keys x 64 d   (padded)
    __shared__ __align__(16) _Float16 lds_v[2][64][72];  // 64 d  x 64 keys  (padded)
    __shared__ __align__(16) float    lds_m[2][128][68]; // 128 q x 64 keys  (padded)

    const int t     = threadIdx.x;
    const int wave  = t >> 5;
    const int lane  = t & 31;
    const int lh    = lane >> 4;     // lane half
    const int ln    = lane & 15;     // lane-in-half == N column (query)
    const int bh    = blockIdx.y;
    const int b     = bh >> 4;       // H = 16
    const int q0blk = blockIdx.x * 128;
    const int qbase = q0blk + wave * 16;

    // issue one 64-key stage (K, V^T, mask tiles) into buffer bufi
    auto issue_stage = [&](int bufi, int kb) {
#pragma unroll
        for (int p = 0; p < 2; ++p) {
            const int id = p * 256 + t, row = id >> 3, seg = id & 7;
            async_cp16(&lds_k[bufi][row][seg * 8],
                       kf + ((size_t)bh * Sc + kb + row) * Dh + seg * 8);
            async_cp16(&lds_v[bufi][row][seg * 8],
                       vt + ((size_t)bh * Dh + row) * Sc + kb + seg * 8);
        }
#pragma unroll
        for (int p = 0; p < 8; ++p) {
            const int id = p * 256 + t, row = id >> 4, c4 = (id & 15) * 4;
            async_cp16(&lds_m[bufi][row][c4],
                       mask + ((size_t)b * Sc + q0blk + row) * Sc + kb + c4);
        }
    };

    // Q fragments as B operand: halves h -> d = 32c + h + 16*lh (contiguous 32B)
    HFrag qfr[2];
    {
        const _Float16* qrow = qf + ((size_t)bh * Sc + qbase + ln) * Dh;
        for (int c = 0; c < 2; ++c) {
            qfr[c].q[0] = *(const uint4*)(qrow + c * 32 + lh * 16);
            qfr[c].q[1] = *(const uint4*)(qrow + c * 32 + lh * 16 + 8);
        }
    }

    v8f acc[4];
    for (int dt = 0; dt < 4; ++dt) acc[dt] = (v8f){0, 0, 0, 0, 0, 0, 0, 0};
    float m = -1e30f, l = 0.0f;

    // prologue: fill buffer 0
    issue_stage(0, 0);
    wait_async();
    __syncthreads();

#pragma unroll 1
    for (int kt = 0; kt < Sc / 64; ++kt) {
        const int cur = kt & 1;
        if (kt + 1 < Sc / 64)
            issue_stage(cur ^ 1, (kt + 1) * 64);   // overlap next fetch with compute

        // ---- S^T = K x Q^T : four 16(key)x16(q) tiles, K-dim 64 in 2 chunks ----
        v8f st[4];
#pragma unroll
        for (int tt = 0; tt < 4; ++tt) {
            const int row = tt * 16 + ln;    // key row (A-matrix M = lane%16)
            HFrag f0, f1;                    // A frags: K=32c+16lh+h (contiguous)
            f0.q[0] = *(const uint4*)&lds_k[cur][row][0 + lh * 16];
            f0.q[1] = *(const uint4*)&lds_k[cur][row][8 + lh * 16];
            f1.q[0] = *(const uint4*)&lds_k[cur][row][32 + lh * 16];
            f1.q[1] = *(const uint4*)&lds_k[cur][row][40 + lh * 16];
            v8f z = (v8f){0, 0, 0, 0, 0, 0, 0, 0};
            st[tt] = wmma_f16(f0, qfr[0], z);
            st[tt] = wmma_f16(f1, qfr[1], st[tt]);
        }

        // ---- additive mask + running max (queries live on lanes) ----
        const int qloc = wave * 16 + ln;
        float mx = -1e30f;
#pragma unroll
        for (int tt = 0; tt < 4; ++tt) {
            float4 m0 = *(const float4*)&lds_m[cur][qloc][tt * 16 + 8 * lh];
            float4 m1 = *(const float4*)&lds_m[cur][qloc][tt * 16 + 8 * lh + 4];
            st[tt][0] += m0.x; st[tt][1] += m0.y; st[tt][2] += m0.z; st[tt][3] += m0.w;
            st[tt][4] += m1.x; st[tt][5] += m1.y; st[tt][6] += m1.z; st[tt][7] += m1.w;
#pragma unroll
            for (int r = 0; r < 8; ++r) mx = fmaxf(mx, st[tt][r]);
        }
        mx = fmaxf(mx, xor16f(mx));
        const float nm = fmaxf(m, mx);
        const float al = __expf(m - nm);

        // ---- exponentiate in place + row sum ----
        float rs = 0.f;
#pragma unroll
        for (int tt = 0; tt < 4; ++tt)
#pragma unroll
            for (int r = 0; r < 8; ++r) {
                st[tt][r] = __expf(st[tt][r] - nm);
                rs += st[tt][r];
            }
        rs += xor16f(rs);
        l = l * al + rs;
        m = nm;

        // ---- P^T as B operand: group g covers keys 32g..32g+31 ----
        // halves h -> key = 32g + h + 16*lh; own regs give one half of the
        // 16 halves, the partner lane (xor 16) gives the other, exchanged as
        // packed f16x2 words via v_permlanex16 (4 per group).
        HFrag pf[2];
#pragma unroll
        for (int g = 0; g < 2; ++g) {
            PPack pa, pb;
#pragma unroll
            for (int j = 0; j < 8; ++j) {
                pa.h[j] = (_Float16)st[2 * g][j];
                pb.h[j] = (_Float16)st[2 * g + 1][j];
            }
            uint4 sel = lh ? pa.q : pb.q;
            uint4 rcv;
            rcv.x = permx16u(sel.x);
            rcv.y = permx16u(sel.y);
            rcv.z = permx16u(sel.z);
            rcv.w = permx16u(sel.w);
            pf[g].q[0] = lh ? rcv : pa.q;    // lh0: own t_even ; lh1: partner t_odd
            pf[g].q[1] = lh ? pb.q : rcv;    // lh0: partner t_even ; lh1: own t_odd
        }

        // ---- O^T += V^T x P^T over 4 d-tiles, 2 key chunks each ----
#pragma unroll
        for (int dt = 0; dt < 4; ++dt) {
            const int drow = dt * 16 + ln;   // d row (A-matrix M)
            HFrag v0, v1;
            v0.q[0] = *(const uint4*)&lds_v[cur][drow][0 + lh * 16];
            v0.q[1] = *(const uint4*)&lds_v[cur][drow][8 + lh * 16];
            v1.q[0] = *(const uint4*)&lds_v[cur][drow][32 + lh * 16];
            v1.q[1] = *(const uint4*)&lds_v[cur][drow][40 + lh * 16];
            v8f c = acc[dt];
#pragma unroll
            for (int r = 0; r < 8; ++r) c[r] *= al;
            c = wmma_f16(v0, pf[0], c);
            acc[dt] = wmma_f16(v1, pf[1], c);
        }

        // my async issues (next tile) must land; then block-wide join
        wait_async();
        __syncthreads();
    }

    // ---- epilogue: C/D layout lane=(q), reg r=(d=16dt+r+8lh) writes directly ----
    const float invl = 1.0f / l;
    float* orow = out + ((size_t)bh * Sc + qbase + ln) * Dh + lh * 8;
#pragma unroll
    for (int dt = 0; dt < 4; ++dt) {
        float4 w0 = make_float4(acc[dt][0] * invl, acc[dt][1] * invl,
                                acc[dt][2] * invl, acc[dt][3] * invl);
        float4 w1 = make_float4(acc[dt][4] * invl, acc[dt][5] * invl,
                                acc[dt][6] * invl, acc[dt][7] * invl);
        *(float4*)(orow + dt * 16)     = w0;
        *(float4*)(orow + dt * 16 + 4) = w1;
    }
}

// ---------------------------------------------------------------------------
extern "C" void kernel_launch(void* const* d_in, const int* in_sizes, int n_in,
                              void* d_out, int out_size, void* d_ws, size_t ws_size,
                              hipStream_t stream) {
    const float* query     = (const float*)d_in[0];
    const float* key       = (const float*)d_in[1];
    const float* value     = (const float*)d_in[2];
    const float* attn_mask = (const float*)d_in[3];
    const float* inv_scale = (const float*)d_in[4];
    const float* Wq        = (const float*)d_in[5];
    const float* bq        = (const float*)d_in[6];
    const float* Wk        = (const float*)d_in[7];
    const float* bk        = (const float*)d_in[8];
    const float* Wv        = (const float*)d_in[9];
    const float* bv        = (const float*)d_in[10];

    const size_t N = (size_t)Bc * Hc * Sc * Dh;   // 4,194,304 elements
    _Float16* qf = (_Float16*)d_ws;
    _Float16* kf = qf + N;
    _Float16* vt = kf + N;

    dim3 pgrid(Sc / 16, Hc, Bc);                  // (128, 16, 2)
    proj_kernel<<<pgrid, 256, 0, stream>>>(query, key, value,
                                           Wq, bq, Wk, bk, Wv, bv, inv_scale,
                                           qf, kf, vt);

    dim3 agrid(Sc / 128, Bc * Hc);                // (16, 32)
    attn_kernel<<<agrid, 256, 0, stream>>>(qf, kf, vt, attn_mask, (float*)d_out);
}
